// QuantumLLM_59004260712623
// MI455X (gfx1250) — compile-verified, matrix operand order checked by reading, not run on hardware
//
#include <hip/hip_runtime.h>
#include <hip/hip_bf16.h>
#include <math.h>

// ---------------- model constants (match reference) ----------------
#define VOCAB 32000
#define DMODEL 512
#define NLAYER 4
#define NHEAD 8
#define SEQ 2048
#define BATCH 2
#define HD (DMODEL / NHEAD)   // 64
#define FFDIM (4 * DMODEL)    // 2048
#define MTOK (BATCH * SEQ)    // 4096

typedef __attribute__((ext_vector_type(16))) __bf16 v16bf;
typedef __attribute__((ext_vector_type(8)))  float  v8f;

union Frag { v16bf v; uint4 q[2]; };

static __device__ inline v8f wmma_bf16(v16bf a, v16bf b, v8f c) {
    // D = A(16x32 bf16) x B(32x16 bf16) + C(16x16 f32)
    return __builtin_amdgcn_wmma_f32_16x16x32_bf16(
        /*neg_a=*/false, a, /*neg_b=*/false, b,
        /*c_mod=*/(short)0, c, /*reuse_a=*/false, /*reuse_b=*/false);
}

// ---- CDNA5 async global->LDS copy (16B per lane), tracked by ASYNCcnt ----
// Generic pointers to LDS carry the LDS byte offset in their low 32 bits
// (aperture encoding, ISA 10.2), so casting a __shared__ address to u32 gives
// the VDST operand for global_load_async_to_lds_*.
static __device__ inline void async_copy_b128(void* lds_ptr, const void* gptr) {
    uint32_t lds_addr = (uint32_t)(uintptr_t)lds_ptr;
    uint64_t gaddr    = (uint64_t)(uintptr_t)gptr;
    asm volatile("global_load_async_to_lds_b128 %0, %1, off"
                 :: "v"(lds_addr), "v"(gaddr)
                 : "memory");
}
static __device__ inline void async_wait0() {
    asm volatile("s_wait_asynccnt 0x0" ::: "memory");
}

// ---------------- cast fp32 -> bf16 ----------------
__global__ void cast_bf16_kernel(const float* __restrict__ in,
                                 __bf16* __restrict__ out, size_t n) {
    size_t i = (size_t)blockIdx.x * blockDim.x + threadIdx.x;
    size_t stride = (size_t)gridDim.x * blockDim.x;
    for (; i < n; i += stride) out[i] = (__bf16)in[i];
}

// ---------------- embedding: h = tok_emb[x] + pos_emb ----------------
__global__ void embed_kernel(const int* __restrict__ x,
                             const float* __restrict__ tok_emb,
                             const float* __restrict__ pos_emb,
                             float* __restrict__ h) {
    int row = blockIdx.x;                 // 0..MTOK-1
    int s   = row % SEQ;
    int tok = x[row];
    int d   = threadIdx.x * 2;            // 256 threads * 2 = 512
    float2 t = *reinterpret_cast<const float2*>(tok_emb + (size_t)tok * DMODEL + d);
    float2 p = *reinterpret_cast<const float2*>(pos_emb + (size_t)s * DMODEL + d);
    float2 r; r.x = t.x + p.x; r.y = t.y + p.y;
    *reinterpret_cast<float2*>(h + (size_t)row * DMODEL + d) = r;
}

// ---------------- layernorm (fp32 in -> bf16 out) ----------------
__global__ __launch_bounds__(256)
void ln_kernel(const float* __restrict__ x, const float* __restrict__ w,
               const float* __restrict__ b, __bf16* __restrict__ out) {
    int row = blockIdx.x;
    int tid = threadIdx.x;
    int d = tid * 2;
    float2 v = *reinterpret_cast<const float2*>(x + (size_t)row * DMODEL + d);
    float sum  = v.x + v.y;
    float sumq = v.x * v.x + v.y * v.y;
    #pragma unroll
    for (int m = 16; m >= 1; m >>= 1) {
        sum  += __shfl_xor(sum,  m, 32);
        sumq += __shfl_xor(sumq, m, 32);
    }
    __shared__ float s1[8], s2[8];
    int wave = tid >> 5, lane = tid & 31;
    if (lane == 0) { s1[wave] = sum; s2[wave] = sumq; }
    __syncthreads();
    float tot = 0.f, totq = 0.f;
    #pragma unroll
    for (int i = 0; i < 8; ++i) { tot += s1[i]; totq += s2[i]; }
    float mean = tot * (1.0f / DMODEL);
    float var  = totq * (1.0f / DMODEL) - mean * mean;
    float rstd = rsqrtf(var + 1e-5f);
    out[(size_t)row * DMODEL + d]     = (__bf16)((v.x - mean) * rstd * w[d]     + b[d]);
    out[(size_t)row * DMODEL + d + 1] = (__bf16)((v.y - mean) * rstd * w[d + 1] + b[d + 1]);
}

// ---------------- superposition: h = h * cos(phase[d] * h) ----------------
__global__ void superpos_kernel(float* __restrict__ h,
                                const float* __restrict__ phase, size_t n) {
    size_t i = (size_t)blockIdx.x * blockDim.x + threadIdx.x;
    size_t stride = (size_t)gridDim.x * blockDim.x;
    for (; i < n; i += stride) {
        float v = h[i];
        h[i] = v * __cosf(phase[i % DMODEL] * v);
    }
}

// ---------------- interference gate: h = hi*sig(hg) + h*(1-sig(hg)) ----------------
__global__ void gate_kernel(float* __restrict__ h, const float* __restrict__ hi,
                            const float* __restrict__ hg, size_t n) {
    size_t i = (size_t)blockIdx.x * blockDim.x + threadIdx.x;
    size_t stride = (size_t)gridDim.x * blockDim.x;
    for (; i < n; i += stride) {
        float g = 1.0f / (1.0f + __expf(-hg[i]));
        h[i] = hi[i] * g + h[i] * (1.0f - g);
    }
}

// ---------------- WMMA GEMM: C = A[M,K](bf16) * W[K,N](bf16) + bias (+res) ----------------
// block tile 128(M) x 64(N), 8 waves (4x2), each wave 32x32 (2x2 wmma tiles), K-step 32.
// A tile staged with async global->LDS (no VGPR round trip); B tile staged through
// VGPRs because it is transposed into B^T layout for contiguous fragment reads.
#define LDA 40
#define LDB 40
__global__ __launch_bounds__(256, 2)
void gemm_bf16_kernel(const __bf16* __restrict__ A, const __bf16* __restrict__ W,
                      const float* __restrict__ bias, const float* __restrict__ res,
                      float* __restrict__ Cf, __bf16* __restrict__ Cb,
                      int M, int N, int K, int act) {
    __shared__ __bf16 sA[128 * LDA];
    __shared__ __bf16 sB[64 * LDB];
    const int tid  = threadIdx.x;
    const int lane = tid & 31;
    const int wave = tid >> 5;
    const int wm   = wave >> 1;          // 0..3
    const int wn   = wave & 1;           // 0..1
    const int half = lane >> 4;
    const int l16  = lane & 15;
    const int m0   = blockIdx.y * 128;
    const int n0   = blockIdx.x * 64;

    v8f acc[2][2];
    #pragma unroll
    for (int i = 0; i < 2; ++i)
        #pragma unroll
        for (int j = 0; j < 2; ++j)
            #pragma unroll
            for (int r = 0; r < 8; ++r) acc[i][j][r] = 0.0f;

    const int arow = tid >> 1;           // 0..127
    const int acol = (tid & 1) * 16;     // 0 or 16
    const int bk   = tid >> 3;           // 0..31
    const int bn   = (tid & 7) * 8;      // 0..56

    // loop-invariant staging pointers (stride per K-step, avoids per-iter 64-bit mul)
    const __bf16* gA = A + (size_t)(m0 + arow) * K + acol;
    const __bf16* gB = W + (size_t)bk * N + n0 + bn;
    const size_t  gBstep = (size_t)32 * N;

    for (int kk = 0; kk < K; kk += 32) {
        // stage A tile 128x32: async global->LDS, 2 x 16B per lane
        async_copy_b128(&sA[arow * LDA + acol],     gA);
        async_copy_b128(&sA[arow * LDA + acol + 8], gA + 8);
        // stage B tile transposed: sB[n][k] = W[kk+k][n0+n]
        union { uint4 q; __bf16 e[8]; } bu;
        bu.q = *reinterpret_cast<const uint4*>(gB);
        #pragma unroll
        for (int j = 0; j < 8; ++j) sB[(bn + j) * LDB + bk] = bu.e[j];
        if (kk + 32 < K)
            __builtin_prefetch(gA + 32, 0, 1);
        gA += 32;
        gB += gBstep;
        async_wait0();
        __syncthreads();

        Frag fa[2], fb[2];
        #pragma unroll
        for (int i = 0; i < 2; ++i) {
            int row = wm * 32 + i * 16 + l16;
            int kb  = half * 8;
            fa[i].q[0] = *reinterpret_cast<const uint4*>(&sA[row * LDA + kb]);
            fa[i].q[1] = *reinterpret_cast<const uint4*>(&sA[row * LDA + kb + 16]);
        }
        #pragma unroll
        for (int j = 0; j < 2; ++j) {
            int nrow = wn * 32 + j * 16 + l16;
            int kb   = half * 8;
            fb[j].q[0] = *reinterpret_cast<const uint4*>(&sB[nrow * LDB + kb]);
            fb[j].q[1] = *reinterpret_cast<const uint4*>(&sB[nrow * LDB + kb + 16]);
        }
        #pragma unroll
        for (int i = 0; i < 2; ++i)
            #pragma unroll
            for (int j = 0; j < 2; ++j)
                acc[i][j] = wmma_bf16(fa[i].v, fb[j].v, acc[i][j]);
        __syncthreads();
    }

    // epilogue: bias (+residual) (+gelu) -> fp32 and/or bf16
    #pragma unroll
    for (int i = 0; i < 2; ++i) {
        #pragma unroll
        for (int j = 0; j < 2; ++j) {
            #pragma unroll
            for (int r = 0; r < 8; ++r) {
                int row = m0 + wm * 32 + i * 16 + r + half * 8;
                int col = n0 + wn * 32 + j * 16 + l16;
                size_t idx = (size_t)row * N + col;
                float v = acc[i][j][r] + bias[col];
                if (res) v += res[idx];
                if (act == 1) v = 0.5f * v * (1.0f + erff(v * 0.70710678118f));
                if (Cf) Cf[idx] = v;
                if (Cb) Cb[idx] = (__bf16)v;
            }
        }
    }
}

// ---------------- flash attention with phase-modulated scores ----------------
// grid: (SEQ/64, NHEAD, BATCH); 128 threads = 4 waves; each wave owns 16 query rows
#define LDK 72
__global__ __launch_bounds__(128, 2)
void attn_kernel(const __bf16* __restrict__ q, const __bf16* __restrict__ k,
                 const __bf16* __restrict__ v, const float* __restrict__ qphase_l,
                 __bf16* __restrict__ o) {
    __shared__ __bf16 sK[64 * LDK];   // [kpos][hd]  (= B^T for Q.K^T)
    __shared__ __bf16 sV[64 * LDK];   // [hd][kpos]  (= B^T for P.V)
    __shared__ __bf16 sP[64 * LDK];   // [qrow][kpos] A-layout for P.V
    const int tid  = threadIdx.x;
    const int lane = tid & 31;
    const int w    = tid >> 5;
    const int half = lane >> 4;
    const int l16  = lane & 15;
    const int qb0  = blockIdx.x * 64;
    const int hh   = blockIdx.y;
    const int b    = blockIdx.z;
    const float qph   = qphase_l[hh];
    const float scale = 0.125f;       // 1/sqrt(64)

    // Q fragments in registers (A-matrix 16x32 x2 over hd)
    Frag fq[2];
    {
        int qrow = qb0 + w * 16 + l16;
        const __bf16* qr = q + (size_t)(b * SEQ + qrow) * DMODEL + hh * HD;
        #pragma unroll
        for (int ks = 0; ks < 2; ++ks) {
            int kb = ks * 32 + half * 8;
            fq[ks].q[0] = *reinterpret_cast<const uint4*>(qr + kb);
            fq[ks].q[1] = *reinterpret_cast<const uint4*>(qr + kb + 16);
        }
    }

    float mrow[8], lrow[8];
    v8f accO[4];
    #pragma unroll
    for (int r = 0; r < 8; ++r) { mrow[r] = -1e30f; lrow[r] = 0.0f; }
    #pragma unroll
    for (int nt = 0; nt < 4; ++nt)
        #pragma unroll
        for (int r = 0; r < 8; ++r) accO[nt][r] = 0.0f;

    const int ntiles = qb0 / 64 + 1;  // causal: only k-blocks <= q-block
    for (int kt = 0; kt < ntiles; ++kt) {
        const int kb0 = kt * 64;
        // ---- stage K (async, layout-preserving) and V (VGPR transpose) ----
        {
            int kpos = tid >> 1;
            int hdb  = (tid & 1) * 32;
            const __bf16* kr = k + (size_t)(b * SEQ + kb0 + kpos) * DMODEL + hh * HD + hdb;
            const __bf16* vr = v + (size_t)(b * SEQ + kb0 + kpos) * DMODEL + hh * HD + hdb;
            #pragma unroll
            for (int c = 0; c < 4; ++c)
                async_copy_b128(&sK[kpos * LDK + hdb + c * 8], kr + c * 8);
            union { uint4 q; __bf16 e[8]; } vu;
            #pragma unroll
            for (int c = 0; c < 4; ++c) {
                vu.q = *reinterpret_cast<const uint4*>(vr + c * 8);
                #pragma unroll
                for (int j = 0; j < 8; ++j)
                    sV[(hdb + c * 8 + j) * LDK + kpos] = vu.e[j];  // transpose
            }
        }
        async_wait0();
        __syncthreads();

        // ---- scores S = Q.K^T (4 n-tiles x 2 k-steps) ----
        v8f sc[4];
        #pragma unroll
        for (int nt = 0; nt < 4; ++nt) {
            #pragma unroll
            for (int r = 0; r < 8; ++r) sc[nt][r] = 0.0f;
            #pragma unroll
            for (int ks = 0; ks < 2; ++ks) {
                Frag fb;
                int nrow = nt * 16 + l16;
                int kb   = ks * 32 + half * 8;
                fb.q[0] = *reinterpret_cast<const uint4*>(&sK[nrow * LDK + kb]);
                fb.q[1] = *reinterpret_cast<const uint4*>(&sK[nrow * LDK + kb + 16]);
                sc[nt] = wmma_bf16(fq[ks].v, fb.v, sc[nt]);
            }
        }

        // ---- phase modulation + causal mask ----
        float pv[4][8];
        float tmax[8];
        #pragma unroll
        for (int r = 0; r < 8; ++r) tmax[r] = -1e30f;
        #pragma unroll
        for (int nt = 0; nt < 4; ++nt) {
            #pragma unroll
            for (int r = 0; r < 8; ++r) {
                int kpos = kb0 + nt * 16 + l16;
                int qpos = qb0 + w * 16 + r + half * 8;
                float s   = sc[nt][r] * scale;
                float val = (kpos <= qpos)
                              ? s * (1.0f + 0.1f * __cosf(qph * s))
                              : -1e30f;
                pv[nt][r] = val;
                tmax[r]   = fmaxf(tmax[r], val);
            }
        }
        #pragma unroll
        for (int m = 1; m <= 8; m <<= 1)
            #pragma unroll
            for (int r = 0; r < 8; ++r)
                tmax[r] = fmaxf(tmax[r], __shfl_xor(tmax[r], m, 32));

        // ---- online softmax update ----
        float fsum[8];
        #pragma unroll
        for (int r = 0; r < 8; ++r) {
            float mnew = fmaxf(mrow[r], tmax[r]);
            float corr = __expf(mrow[r] - mnew);
            mrow[r] = mnew;
            lrow[r] *= corr;
            #pragma unroll
            for (int nt = 0; nt < 4; ++nt) accO[nt][r] *= corr;
            fsum[r] = 0.0f;
        }
        #pragma unroll
        for (int nt = 0; nt < 4; ++nt)
            #pragma unroll
            for (int r = 0; r < 8; ++r) {
                float p = __expf(pv[nt][r] - mrow[r]);
                pv[nt][r] = p;
                fsum[r] += p;
            }
        #pragma unroll
        for (int m = 1; m <= 8; m <<= 1)
            #pragma unroll
            for (int r = 0; r < 8; ++r)
                fsum[r] += __shfl_xor(fsum[r], m, 32);
        #pragma unroll
        for (int r = 0; r < 8; ++r) lrow[r] += fsum[r];

        // ---- P -> LDS (accumulator layout -> A-fragment layout) ----
        #pragma unroll
        for (int nt = 0; nt < 4; ++nt)
            #pragma unroll
            for (int r = 0; r < 8; ++r)
                sP[(w * 16 + r + half * 8) * LDK + nt * 16 + l16] = (__bf16)pv[nt][r];
        __syncthreads();

        // ---- O += P.V ----
        #pragma unroll
        for (int ks = 0; ks < 2; ++ks) {
            Frag fp;
            int prow = w * 16 + l16;
            int kb   = ks * 32 + half * 8;
            fp.q[0] = *reinterpret_cast<const uint4*>(&sP[prow * LDK + kb]);
            fp.q[1] = *reinterpret_cast<const uint4*>(&sP[prow * LDK + kb + 16]);
            #pragma unroll
            for (int nt = 0; nt < 4; ++nt) {
                Frag fv;
                int nrow = nt * 16 + l16;
                fv.q[0] = *reinterpret_cast<const uint4*>(&sV[nrow * LDK + kb]);
                fv.q[1] = *reinterpret_cast<const uint4*>(&sV[nrow * LDK + kb + 16]);
                accO[nt] = wmma_bf16(fp.v, fv.v, accO[nt]);
            }
        }
        __syncthreads();
    }

    // ---- epilogue: O / l ----
    #pragma unroll
    for (int nt = 0; nt < 4; ++nt)
        #pragma unroll
        for (int r = 0; r < 8; ++r) {
            int qpos = qb0 + w * 16 + r + half * 8;
            int col  = hh * HD + nt * 16 + l16;
            o[(size_t)(b * SEQ + qpos) * DMODEL + col] =
                (__bf16)(accO[nt][r] / lrow[r]);
        }
}

// ---------------- host launcher ----------------
extern "C" void kernel_launch(void* const* d_in, const int* in_sizes, int n_in,
                              void* d_out, int out_size, void* d_ws, size_t ws_size,
                              hipStream_t stream) {
    (void)in_sizes; (void)n_in; (void)out_size; (void)ws_size;

    const int*   x        = (const int*)d_in[0];
    const float* tok_emb  = (const float*)d_in[1];
    const float* pos_emb  = (const float*)d_in[2];
    const float* qw = (const float*)d_in[3];  const float* qb = (const float*)d_in[4];
    const float* kw = (const float*)d_in[5];  const float* kb = (const float*)d_in[6];
    const float* vw = (const float*)d_in[7];  const float* vb = (const float*)d_in[8];
    const float* ow = (const float*)d_in[9];  const float* ob = (const float*)d_in[10];
    const float* qphase   = (const float*)d_in[11];
    const float* sp_phase = (const float*)d_in[12];
    const float* n1w = (const float*)d_in[13]; const float* n1b = (const float*)d_in[14];
    const float* n2w = (const float*)d_in[15]; const float* n2b = (const float*)d_in[16];
    const float* f1w = (const float*)d_in[17]; const float* f1b = (const float*)d_in[18];
    const float* f2w = (const float*)d_in[19]; const float* f2b = (const float*)d_in[20];
    const float* iw  = (const float*)d_in[21]; const float* ib  = (const float*)d_in[22];
    const float* gw  = (const float*)d_in[23]; const float* gb  = (const float*)d_in[24];
    const float* nw  = (const float*)d_in[25]; const float* nb  = (const float*)d_in[26];
    const float* outw = (const float*)d_in[27]; const float* outb = (const float*)d_in[28];
    float* logits = (float*)d_out;

    // workspace carve-up
    char* p = (char*)d_ws;
    auto alloc = [&](size_t bytes) -> void* {
        void* r = (void*)p;
        p += (bytes + 255) & ~(size_t)255;
        return r;
    };
    const size_t WDD = (size_t)NLAYER * DMODEL * DMODEL;
    const size_t WDF = (size_t)NLAYER * DMODEL * FFDIM;
    __bf16* wq_b = (__bf16*)alloc(WDD * 2);
    __bf16* wk_b = (__bf16*)alloc(WDD * 2);
    __bf16* wv_b = (__bf16*)alloc(WDD * 2);
    __bf16* wo_b = (__bf16*)alloc(WDD * 2);
    __bf16* wi_b = (__bf16*)alloc(WDD * 2);
    __bf16* wg_b = (__bf16*)alloc(WDD * 2);
    __bf16* wf1_b = (__bf16*)alloc(WDF * 2);
    __bf16* wf2_b = (__bf16*)alloc(WDF * 2);
    __bf16* wout_b = (__bf16*)alloc((size_t)DMODEL * VOCAB * 2);
    float*  h     = (float*)alloc((size_t)MTOK * DMODEL * 4);
    __bf16* hn_b  = (__bf16*)alloc((size_t)MTOK * DMODEL * 2);
    __bf16* h_b   = (__bf16*)alloc((size_t)MTOK * DMODEL * 2);
    __bf16* q_b   = (__bf16*)alloc((size_t)MTOK * DMODEL * 2);
    __bf16* k_b   = (__bf16*)alloc((size_t)MTOK * DMODEL * 2);
    __bf16* v_b   = (__bf16*)alloc((size_t)MTOK * DMODEL * 2);
    __bf16* o_b   = (__bf16*)alloc((size_t)MTOK * DMODEL * 2);
    __bf16* ff1_b = (__bf16*)alloc((size_t)MTOK * FFDIM * 2);
    float*  hi    = (float*)alloc((size_t)MTOK * DMODEL * 4);
    float*  hg    = (float*)alloc((size_t)MTOK * DMODEL * 4);

    auto cast = [&](const float* src, __bf16* dst, size_t n) {
        int blocks = (int)((n + 1023) / 1024);
        if (blocks > 2048) blocks = 2048;
        cast_bf16_kernel<<<blocks, 256, 0, stream>>>(src, dst, n);
    };
    cast(qw, wq_b, WDD);  cast(kw, wk_b, WDD);  cast(vw, wv_b, WDD);
    cast(ow, wo_b, WDD);  cast(iw, wi_b, WDD);  cast(gw, wg_b, WDD);
    cast(f1w, wf1_b, WDF); cast(f2w, wf2_b, WDF);
    cast(outw, wout_b, (size_t)DMODEL * VOCAB);

    auto gemm = [&](const __bf16* A, const __bf16* W, const float* bias,
                    const float* res, float* Cf, __bf16* Cb,
                    int M, int N, int K, int act) {
        dim3 g(N / 64, M / 128);
        gemm_bf16_kernel<<<g, 256, 0, stream>>>(A, W, bias, res, Cf, Cb, M, N, K, act);
    };

    // embedding
    embed_kernel<<<MTOK, 256, 0, stream>>>(x, tok_emb, pos_emb, h);

    const size_t HN = (size_t)MTOK * DMODEL;
    for (int l = 0; l < NLAYER; ++l) {
        const size_t odd = (size_t)l * DMODEL * DMODEL;
        const size_t odf = (size_t)l * DMODEL * FFDIM;
        const size_t od  = (size_t)l * DMODEL;
        const size_t off = (size_t)l * FFDIM;

        // attention
        ln_kernel<<<MTOK, 256, 0, stream>>>(h, n1w + od, n1b + od, hn_b);
        gemm(hn_b, wq_b + odd, qb + od, nullptr, nullptr, q_b, MTOK, DMODEL, DMODEL, 0);
        gemm(hn_b, wk_b + odd, kb + od, nullptr, nullptr, k_b, MTOK, DMODEL, DMODEL, 0);
        gemm(hn_b, wv_b + odd, vb + od, nullptr, nullptr, v_b, MTOK, DMODEL, DMODEL, 0);
        dim3 ag(SEQ / 64, NHEAD, BATCH);
        attn_kernel<<<ag, 128, 0, stream>>>(q_b, k_b, v_b, qphase + (size_t)l * NHEAD, o_b);
        gemm(o_b, wo_b + odd, ob + od, h, h, nullptr, MTOK, DMODEL, DMODEL, 0);

        // superposition
        superpos_kernel<<<2048, 256, 0, stream>>>(h, sp_phase + od, HN);

        // feed-forward (exact GELU fused into first GEMM)
        ln_kernel<<<MTOK, 256, 0, stream>>>(h, n2w + od, n2b + od, hn_b);
        gemm(hn_b, wf1_b + odf, f1b + off, nullptr, nullptr, ff1_b, MTOK, FFDIM, DMODEL, 1);
        gemm(ff1_b, wf2_b + odf, f2b + od, h, h, h_b, MTOK, DMODEL, FFDIM, 0);

        // interference gate
        gemm(h_b, wi_b + odd, ib + od, nullptr, hi, nullptr, MTOK, DMODEL, DMODEL, 0);
        gemm(h_b, wg_b + odd, gb + od, nullptr, hg, nullptr, MTOK, DMODEL, DMODEL, 0);
        gate_kernel<<<2048, 256, 0, stream>>>(h, hi, hg, HN);
    }

    // final LN + logits
    ln_kernel<<<MTOK, 256, 0, stream>>>(h, nw, nb, hn_b);
    gemm(hn_b, wout_b, outb, nullptr, logits, nullptr, MTOK, VOCAB, DMODEL, 0);
}
